// MinLSTM_91070486544682
// MI455X (gfx1250) — matrix operand order repaired and checked by reading, not compiled, not run on hardware
//
#include <hip/hip_runtime.h>
#include <hip/hip_bf16.h>

// ---------------------------------------------------------------------------
// MinLSTM on MI455X (gfx1250, wave32):
//   phase 0: cast x and W_{f,i,h} to f16 (halves GEMM traffic; WMMA path)
//   phase 1: fused 3-gate GEMM via v_wmma_f32_16x16x32_f16.
//            Each wave: 32x16 output tile (2 M-subtiles share B fragments),
//            register double-buffered K-loop so loads overlap WMMAs.
//            Epilogue computes log-gate math -> log_f in ws, log_v in d_out.
//   phase 2: chunked associative scan over T in log space:
//            (A1,V1)o(A2,V2) = (A1+A2, logaddexp(V1+A2, V2))
//     2a: per-chunk aggregate (64 chunks x 64 steps, coalesced over H)
//     2b: scan of 64 chunk aggregates per (b,h), seeded with log_g(h0)
//     2c: replay chunks with prefix state, write exp(log_h) into d_out
// ---------------------------------------------------------------------------

typedef _Float16 v16h __attribute__((ext_vector_type(16)));
typedef _Float16 v8h  __attribute__((ext_vector_type(8)));
typedef _Float16 v4h  __attribute__((ext_vector_type(4)));
typedef float    v8f  __attribute__((ext_vector_type(8)));

#define B_  8
#define T_  4096
#define D_  512
#define H_  512
#define M_  (B_ * T_)       // 32768 rows
#define NC_ 64              // chunks along T
#define TC_ (T_ / NC_)      // 64 steps per chunk

__device__ __forceinline__ float softplus_f(float x) {
    // stable softplus: max(x,0) + log1p(exp(-|x|))
    return fmaxf(x, 0.f) + log1pf(expf(-fabsf(x)));
}

__device__ __forceinline__ float logaddexp_f(float a, float b) {
    float m = fmaxf(a, b);
    if (m == -__builtin_inff()) return -__builtin_inff();
    return m + log1pf(expf(fminf(a, b) - m));
}

__device__ __forceinline__ float log_g_f(float x) {
    // log g(x); g(x) = x + 0.5 if x >= 0 else sigmoid(x)
    return (x >= 0.f) ? logf(x + 0.5f) : -softplus_f(-x);
}

// A-fragment gather: per lane, 16 contiguous halves at
// k = kb + sel*16 .. kb + sel*16 + 15 (sel folded into ptr).
__device__ __forceinline__ v16h load_afrag(const _Float16* p, int kb) {
    v8h lo = *(const v8h*)(p + kb);
    v8h hi = *(const v8h*)(p + kb + 8);
    return __builtin_shufflevector(lo, hi,
               0, 1, 2, 3, 4, 5, 6, 7, 8, 9, 10, 11, 12, 13, 14, 15);
}

#define WMMA_F16(A, Bm, C) \
    __builtin_amdgcn_wmma_f32_16x16x32_f16(false, (A), false, (Bm), (short)0, (C), false, false)

// ---------------------------------------------------------------- phase 0 ---
__global__ void __launch_bounds__(256) cast_f32_to_f16(
    const float* __restrict__ src, _Float16* __restrict__ dst, int n)
{
    int i = (blockIdx.x * blockDim.x + threadIdx.x) * 4;
    if (i + 3 < n) {
        float4 v = *(const float4*)(src + i);
        v4h o;
        o[0] = (_Float16)v.x; o[1] = (_Float16)v.y;
        o[2] = (_Float16)v.z; o[3] = (_Float16)v.w;
        *(v4h*)(dst + i) = o;
    }
}

// ---------------------------------------------------------------- phase 1 ---
__global__ void __launch_bounds__(256) gemm_gates(
    const _Float16* __restrict__ xh,   // [M, D] f16
    const _Float16* __restrict__ wf,   // [H, D] f16 (row n = B column n)
    const _Float16* __restrict__ wi,
    const _Float16* __restrict__ wh,
    const float* __restrict__ b_f,
    const float* __restrict__ b_i,
    const float* __restrict__ b_h,
    float* __restrict__ lf_out,        // [M, H] log_f
    float* __restrict__ lv_out)        // [M, H] log_v (lives in d_out)
{
    const int lane   = threadIdx.x & 31;
    const int wave   = (blockIdx.x * blockDim.x + threadIdx.x) >> 5;
    const int tile_m = wave >> 5;       // M/32 = 1024 tiles of 32 rows
    const int tile_n = wave & 31;       // N/16 = 32 tiles

    const int row = lane & 15;          // A: M row; B: N column
    const int sel = lane >> 4;          // half-wave select

    const _Float16* aptr0 = xh + (size_t)(tile_m * 32 + row) * D_ + sel * 16;
    const _Float16* aptr1 = aptr0 + (size_t)16 * D_;
    // B fragment: lane holds column n = tile_n*16+row; contiguous 16 halves
    // at k = kb + sel*16 (B[k,n] = W[n,k], W kept in [H,D] row-major).
    const size_t boff = (size_t)(tile_n * 16 + row) * D_ + sel * 16;
    const _Float16* pf = wf + boff;
    const _Float16* pi = wi + boff;
    const _Float16* ph = wh + boff;

    v8f accf0 = {}, acci0 = {}, acch0 = {};
    v8f accf1 = {}, acci1 = {}, acch1 = {};

    // Software pipeline: stage 0 loads, loop prefetches next K-block before
    // issuing this block's 6 WMMAs (loads overlap matrix ops).
    v16h A0c = load_afrag(aptr0, 0);
    v16h A1c = load_afrag(aptr1, 0);
    v16h Bfc = *(const v16h*)(pf);
    v16h Bic = *(const v16h*)(pi);
    v16h Bhc = *(const v16h*)(ph);

#pragma unroll
    for (int kb = 0; kb < D_; kb += 32) {
        v16h A0n = A0c, A1n = A1c, Bfn = Bfc, Bin = Bic, Bhn = Bhc;
        if (kb + 32 < D_) {
            A0n = load_afrag(aptr0, kb + 32);
            A1n = load_afrag(aptr1, kb + 32);
            Bfn = *(const v16h*)(pf + kb + 32);
            Bin = *(const v16h*)(pi + kb + 32);
            Bhn = *(const v16h*)(ph + kb + 32);
        }
        accf0 = WMMA_F16(A0c, Bfc, accf0);
        accf1 = WMMA_F16(A1c, Bfc, accf1);
        acci0 = WMMA_F16(A0c, Bic, acci0);
        acci1 = WMMA_F16(A1c, Bic, acci1);
        acch0 = WMMA_F16(A0c, Bhc, acch0);
        acch1 = WMMA_F16(A1c, Bhc, acch1);
        A0c = A0n; A1c = A1n; Bfc = Bfn; Bic = Bin; Bhc = Bhn;
    }

    // Epilogue: C/D layout -> element r is (M = msub + r + 8*sel,
    //                                       N = tile_n*16 + (lane & 15)).
    const int n = tile_n * 16 + (lane & 15);
    const float bfs = b_f[n], bis = b_i[n], bhs = b_h[n];

#pragma unroll
    for (int half = 0; half < 2; ++half) {
        const v8f& af = half ? accf1 : accf0;
        const v8f& ai = half ? acci1 : acci0;
        const v8f& ah = half ? acch1 : acch0;
        int mbase = tile_m * 32 + half * 16 + sel * 8;
#pragma unroll
        for (int r = 0; r < 8; ++r) {
            int m = mbase + r;
            float zf = af[r] + bfs;
            float zi = ai[r] + bis;
            float zh = ah[r] + bhs;
            float diff = softplus_f(-zf) - softplus_f(-zi);
            float lf = -softplus_f(diff);      // log f_t (normalized)
            float li = -softplus_f(-diff);     // log i_t (normalized)
            float lv = li + log_g_f(zh);       // log v_t
            size_t o = (size_t)m * H_ + n;
            lf_out[o] = lf;
            lv_out[o] = lv;
        }
    }
}

// --------------------------------------------------------------- phase 2a ---
__global__ void __launch_bounds__(512) scan_chunk_reduce(
    const float* __restrict__ lf_buf, const float* __restrict__ lv_buf,
    float* __restrict__ aggA, float* __restrict__ aggV)
{
    int b = blockIdx.x / NC_;
    int c = blockIdx.x % NC_;
    int h = threadIdx.x;
    float A = 0.f, V = -__builtin_inff();
    size_t base = ((size_t)b * T_ + (size_t)c * TC_) * H_ + h;
    for (int i = 0; i < TC_; ++i) {
        float lf = lf_buf[base + (size_t)i * H_];
        float lv = lv_buf[base + (size_t)i * H_];
        V = logaddexp_f(V + lf, lv);
        A += lf;
    }
    size_t o = ((size_t)b * NC_ + c) * H_ + h;
    aggA[o] = A;
    aggV[o] = V;
}

// --------------------------------------------------------------- phase 2b ---
__global__ void __launch_bounds__(512) scan_chunk_scan(
    const float* __restrict__ aggA, const float* __restrict__ aggV,
    const float* __restrict__ h0, float* __restrict__ prefix)
{
    int b = blockIdx.x;
    int h = threadIdx.x;
    float s = log_g_f(h0[b * H_ + h]);   // state entering chunk 0
    prefix[((size_t)b * NC_) * H_ + h] = s;
    for (int c = 0; c < NC_ - 1; ++c) {
        size_t o = ((size_t)b * NC_ + c) * H_ + h;
        s = logaddexp_f(s + aggA[o], aggV[o]);
        prefix[((size_t)b * NC_ + c + 1) * H_ + h] = s;
    }
}

// --------------------------------------------------------------- phase 2c ---
__global__ void __launch_bounds__(512) scan_apply(
    const float* __restrict__ lf_buf,
    float* __restrict__ out_lv,          // in: log_v, out: exp(log_h)
    const float* __restrict__ prefix)
{
    int b = blockIdx.x / NC_;
    int c = blockIdx.x % NC_;
    int h = threadIdx.x;
    float s = prefix[((size_t)b * NC_ + c) * H_ + h];
    size_t base = ((size_t)b * T_ + (size_t)c * TC_) * H_ + h;
    for (int i = 0; i < TC_; ++i) {
        size_t o = base + (size_t)i * H_;
        float lf = lf_buf[o];
        float lv = out_lv[o];
        s = logaddexp_f(s + lf, lv);
        out_lv[o] = expf(s);
    }
}

// ---------------------------------------------------------------------------
extern "C" void kernel_launch(void* const* d_in, const int* in_sizes, int n_in,
                              void* d_out, int out_size, void* d_ws, size_t ws_size,
                              hipStream_t stream)
{
    const float* x   = (const float*)d_in[0];
    const float* h0  = (const float*)d_in[1];
    const float* W_f = (const float*)d_in[2];
    const float* b_f = (const float*)d_in[3];
    const float* W_i = (const float*)d_in[4];
    const float* b_i = (const float*)d_in[5];
    const float* W_h = (const float*)d_in[6];
    const float* b_h = (const float*)d_in[7];
    float* out = (float*)d_out;

    // workspace carve-up (~100.5 MB)
    char* p = (char*)d_ws;
    _Float16* xh = (_Float16*)p;  p += (size_t)M_ * D_ * sizeof(_Float16);
    _Float16* wf = (_Float16*)p;  p += (size_t)H_ * D_ * sizeof(_Float16);
    _Float16* wi = (_Float16*)p;  p += (size_t)H_ * D_ * sizeof(_Float16);
    _Float16* wh = (_Float16*)p;  p += (size_t)H_ * D_ * sizeof(_Float16);
    float* lfbuf  = (float*)p;    p += (size_t)M_ * H_ * sizeof(float);
    float* aggA   = (float*)p;    p += (size_t)B_ * NC_ * H_ * sizeof(float);
    float* aggV   = (float*)p;    p += (size_t)B_ * NC_ * H_ * sizeof(float);
    float* prefix = (float*)p;

    // phase 0: f32 -> f16 casts
    cast_f32_to_f16<<<(M_ * D_ / 4 + 255) / 256, 256, 0, stream>>>(x, xh, M_ * D_);
    cast_f32_to_f16<<<(H_ * D_ / 4 + 255) / 256, 256, 0, stream>>>(W_f, wf, H_ * D_);
    cast_f32_to_f16<<<(H_ * D_ / 4 + 255) / 256, 256, 0, stream>>>(W_i, wi, H_ * D_);
    cast_f32_to_f16<<<(H_ * D_ / 4 + 255) / 256, 256, 0, stream>>>(W_h, wh, H_ * D_);

    // phase 1: fused 3-gate WMMA GEMM; 32768 waves (32x16 tile each)
    gemm_gates<<<(M_ / 32) * (H_ / 16) / 8, 256, 0, stream>>>(
        xh, wf, wi, wh, b_f, b_i, b_h, lfbuf, out);

    // phase 2: chunked log-space associative scan over T
    scan_chunk_reduce<<<B_ * NC_, H_, 0, stream>>>(lfbuf, out, aggA, aggV);
    scan_chunk_scan<<<B_, H_, 0, stream>>>(aggA, aggV, h0, prefix);
    scan_apply<<<B_ * NC_, H_, 0, stream>>>(lfbuf, out, prefix);
}
